// ISSIMLoss_32581621907914
// MI455X (gfx1250) — matrix operand order repaired and checked by reading, not compiled, not run on hardware
//
#include <hip/hip_runtime.h>

// ---------------------------------------------------------------------------
// ISSIM loss on MI455X (gfx1250).
// Each 33x33 "same" depthwise Gaussian conv on a 32x32 crop is exactly
//   out = A * X * A,  A[p][q] = g(p-q)  (symmetric 32x32 band matrix),
// so each conv = two 32x32x32 matmuls = 8 x v_wmma_f32_16x16x32_f16.
// One wave (32 threads) per crop; 3200 crops; 40 WMMAs per crop.
// ---------------------------------------------------------------------------

typedef __attribute__((ext_vector_type(16))) _Float16 v16h;
typedef __attribute__((ext_vector_type(8)))  float    v8f;

#define NB      32          // B
#define NP      100         // points per image
#define IMH     1024
#define IMW     1024
#define OFFR    16          // LOCAL//2
#define CS      32          // crop size
#define LSTR    33          // padded LDS row stride (elements)
#define C1_     (0.01f * 0.01f)
#define C2_     (0.03f * 0.03f)

__device__ __forceinline__ float gauss_unnorm(int d) {
    // exp(-d^2 / (2 * 1.5^2))
    return __expf(-(float)(d * d) * (1.0f / 4.5f));
}

__device__ __forceinline__ v8f wmma_f16(v16h a, v16h b, v8f c) {
    // D(f32, 16x16) = A(f16, 16x32) * B(f16, 32x16) + C
    return __builtin_amdgcn_wmma_f32_16x16x32_f16(
        /*neg_a=*/false, a, /*neg_b=*/false, b,
        /*c_mod=*/(short)0, c, /*reuse_a=*/false, /*reuse_b=*/false);
}

struct GaussFrags {
    v16h aa[2];   // A as WMMA A-fragment, row blocks m in [0,16) and [16,32)
    v16h ab[2];   // A as WMMA B-fragment, col blocks n in [0,16) and [16,32)
};

// A-fragment element index (contiguous k order, matching B fragments):
// lane L (grp=L>>4, idx=L&15) holds row m = rowBase+idx,
// half h -> K = 16*grp + h.
__device__ __forceinline__ int afrag_k(int h, int grp) {
    return 16 * grp + h;
}

__device__ __forceinline__ void build_gauss_frags(int lane, GaussFrags& g) {
    float Z = 0.0f;
#pragma unroll
    for (int d = -16; d <= 16; ++d) Z += gauss_unnorm(d);
    const float invZ = 1.0f / Z;
    const int grp = lane >> 4;
    const int idx = lane & 15;
#pragma unroll
    for (int blk = 0; blk < 2; ++blk) {
#pragma unroll
        for (int h = 0; h < 16; ++h) {
            // A-fragment: element A[m][k]
            int k = afrag_k(h, grp);
            int m = 16 * blk + idx;
            int d1 = m - k;
            float va = (d1 >= -16 && d1 <= 16) ? gauss_unnorm(d1) * invZ : 0.0f;
            g.aa[blk][h] = (_Float16)va;
            // B-fragment: element A[K][n], K = 16*grp + h (row striped across lanes)
            int kk = 16 * grp + h;
            int n  = 16 * blk + idx;
            int d2 = kk - n;
            float vb = (d2 >= -16 && d2 <= 16) ? gauss_unnorm(d2) * invZ : 0.0f;
            g.ab[blk][h] = (_Float16)vb;
        }
    }
}

// Build X as a WMMA A-fragment for one of the 5 variants, fused product.
// OPV: 0=t, 1=o, 2=t*t, 3=o*o, 4=t*o
template <int OPV>
__device__ __forceinline__ v16h build_x_afrag(const float* tS, const float* oS,
                                              int lane, int rowBase) {
    const int grp = lane >> 4;
    const int m = rowBase + (lane & 15);
    v16h a;
#pragma unroll
    for (int h = 0; h < 16; ++h) {
        int k = afrag_k(h, grp);
        float tv = tS[m * LSTR + k];
        float ov = oS[m * LSTR + k];
        float x = (OPV == 0) ? tv
                : (OPV == 1) ? ov
                : (OPV == 2) ? tv * tv
                : (OPV == 3) ? ov * ov
                             : tv * ov;
        a[h] = (_Float16)x;
    }
    return a;
}

// Store a 16x16 f32 D tile to an f16 LDS matrix at (tm, tn).
// D layout: lane L holds n = (L&15), rows m = 8*(L>>4) + r, r = 0..7.
__device__ __forceinline__ void store_tile_f16(_Float16* Y, int lane, int tm,
                                               int tn, v8f d) {
    const int grp = lane >> 4;
    const int n = tn + (lane & 15);
#pragma unroll
    for (int r = 0; r < 8; ++r) {
        int m = tm + 8 * grp + r;
        Y[m * LSTR + n] = (_Float16)d[r];
    }
}

// Build a WMMA B-fragment (32x16) from an f16 LDS matrix, cols [colBase, +16).
__device__ __forceinline__ v16h build_bfrag_f16(const _Float16* Y, int lane,
                                                int colBase) {
    const int grp = lane >> 4;
    const int n = colBase + (lane & 15);
    v16h b;
#pragma unroll
    for (int h = 0; h < 16; ++h) {
        int k = 16 * grp + h;
        b[h] = Y[k * LSTR + n];
    }
    return b;
}

template <int OPV>
__device__ __forceinline__ void stage1(const float* tS, const float* oS,
                                       _Float16* Y, int lane,
                                       const GaussFrags& g) {
    v16h x0 = build_x_afrag<OPV>(tS, oS, lane, 0);
    v16h x1 = build_x_afrag<OPV>(tS, oS, lane, 16);
    v8f z = {};
    v8f y00 = wmma_f16(x0, g.ab[0], z);
    v8f y01 = wmma_f16(x0, g.ab[1], z);
    v8f y10 = wmma_f16(x1, g.ab[0], z);
    v8f y11 = wmma_f16(x1, g.ab[1], z);
    store_tile_f16(Y, lane, 0, 0, y00);
    store_tile_f16(Y, lane, 0, 16, y01);
    store_tile_f16(Y, lane, 16, 0, y10);
    store_tile_f16(Y, lane, 16, 16, y11);
}

__global__ __launch_bounds__(32) void issim_crop_kernel(
    const float* __restrict__ out_img, const float* __restrict__ tgt_img,
    const int* __restrict__ points, float* __restrict__ crop_sums) {
    __shared__ float tS[CS * LSTR];
    __shared__ float oS[CS * LSTR];
    __shared__ _Float16 Yh[5][CS * LSTR];

    const int cid = blockIdx.x;         // crop id in [0, NB*NP)
    const int lane = threadIdx.x;       // wave32
    const int b = cid / NP;
    const int y = points[cid * 2 + 0];
    const int x = points[cid * 2 + 1];

    const size_t base = (size_t)b * ((size_t)IMH * IMW) +
                        (size_t)(y - OFFR) * IMW + (size_t)(x - OFFR);
    const float* tp = tgt_img + base;
    const float* op = out_img + base;
#pragma unroll 4
    for (int r = 0; r < CS; ++r) {
        tS[r * LSTR + lane] = tp[(size_t)r * IMW + lane];
        oS[r * LSTR + lane] = op[(size_t)r * IMW + lane];
    }
    __syncthreads();

    GaussFrags g;
    build_gauss_frags(lane, g);

    // Stage 1: Y_V = X_V * A  (five variants, f32 accum -> f16 LDS)
    stage1<0>(tS, oS, Yh[0], lane, g);   // t
    stage1<1>(tS, oS, Yh[1], lane, g);   // o
    stage1<2>(tS, oS, Yh[2], lane, g);   // t*t
    stage1<3>(tS, oS, Yh[3], lane, g);   // o*o
    stage1<4>(tS, oS, Yh[4], lane, g);   // t*o
    __syncthreads();

    // Stage 2: out_V = A * Y_V, tile by tile; combine elementwise into SSIM.
    float local = 0.0f;
#pragma unroll
    for (int I = 0; I < 2; ++I) {
#pragma unroll
        for (int J = 0; J < 2; ++J) {
            v8f z = {};
            v16h bT  = build_bfrag_f16(Yh[0], lane, 16 * J);
            v16h bO  = build_bfrag_f16(Yh[1], lane, 16 * J);
            v16h bTT = build_bfrag_f16(Yh[2], lane, 16 * J);
            v16h bOO = build_bfrag_f16(Yh[3], lane, 16 * J);
            v16h bTO = build_bfrag_f16(Yh[4], lane, 16 * J);
            v8f mu1 = wmma_f16(g.aa[I], bT, z);
            v8f mu2 = wmma_f16(g.aa[I], bO, z);
            v8f ett = wmma_f16(g.aa[I], bTT, z);
            v8f eoo = wmma_f16(g.aa[I], bOO, z);
            v8f eto = wmma_f16(g.aa[I], bTO, z);
#pragma unroll
            for (int r = 0; r < 8; ++r) {
                float m1 = mu1[r], m2 = mu2[r];
                float m1s = m1 * m1, m2s = m2 * m2, m12 = m1 * m2;
                float s1  = ett[r] - m1s;
                float s2  = eoo[r] - m2s;
                float s12 = eto[r] - m12;
                float num = (2.0f * m12 + C1_) * (2.0f * s12 + C2_);
                float den = (m1s + m2s + C1_) * (s1 + s2 + C2_);
                local += 1.0f - num / den;
            }
        }
    }

    // Deterministic wave32 butterfly reduction.
#pragma unroll
    for (int m = 16; m >= 1; m >>= 1) local += __shfl_xor(local, m, 32);
    if (lane == 0) crop_sums[cid] = local;
}

__global__ __launch_bounds__(256) void issim_reduce_kernel(
    const float* __restrict__ s, float* __restrict__ out, int n) {
    __shared__ float buf[256];
    float acc = 0.0f;
    for (int i = threadIdx.x; i < n; i += 256) acc += s[i];
    buf[threadIdx.x] = acc;
    __syncthreads();
    for (int w = 128; w > 0; w >>= 1) {
        if ((int)threadIdx.x < w) buf[threadIdx.x] += buf[threadIdx.x + w];
        __syncthreads();
    }
    if (threadIdx.x == 0) out[0] = buf[0] * (1.0f / (float)(NB * NP));
}

extern "C" void kernel_launch(void* const* d_in, const int* in_sizes, int n_in,
                              void* d_out, int out_size, void* d_ws,
                              size_t ws_size, hipStream_t stream) {
    const float* out_img = (const float*)d_in[0];   // "output"
    const float* tgt_img = (const float*)d_in[1];   // "target"
    const int*   points  = (const int*)d_in[2];     // [B, P, 2] (y, x)
    float* crop_sums = (float*)d_ws;                // NB*NP floats of scratch

    issim_crop_kernel<<<NB * NP, 32, 0, stream>>>(out_img, tgt_img, points,
                                                  crop_sums);
    issim_reduce_kernel<<<1, 256, 0, stream>>>(crop_sums, (float*)d_out,
                                               NB * NP);
}
